// STnonlocalGNLStage_48584670052544
// MI455X (gfx1250) — matrix operand order repaired and checked by reading, not compiled, not run on hardware
//
#include <hip/hip_runtime.h>
#include <hip/hip_bf16.h>

// ---------------------------------------------------------------------------
// Problem dims (fixed by the reference)
// ---------------------------------------------------------------------------
constexpr int Bc  = 8;            // batch (clips)
constexpr int SEQ = 8;            // frames
constexpr int C   = 1024;         // inplanes
constexpr int Pd  = 512;          // planes
constexpr int H   = 14, W = 14;
constexpr int HW  = H * W;        // 196
constexpr int L   = SEQ * HW;     // 1568
constexpr int BN  = Bc * SEQ;     // 64
constexpr float BN_EPS = 1e-5f;

typedef __attribute__((ext_vector_type(16))) __bf16          v16bf;
typedef __attribute__((ext_vector_type(8)))  __bf16          v8bf;
typedef __attribute__((ext_vector_type(8)))  float           v8f;
typedef __attribute__((ext_vector_type(8)))  unsigned short  v8u16;
typedef __attribute__((ext_vector_type(4)))  int             v4i;

#if defined(__AMDGCN__) && __has_builtin(__builtin_amdgcn_global_load_async_to_lds_b128) && \
    __has_builtin(__builtin_amdgcn_s_wait_asynccnt)
#define USE_ASYNC_LDS 1
// builtin signature (from hipcc diagnostic): (v4i AS1* src, v4i AS3* dst, imm offset, imm cpol)
typedef __attribute__((address_space(1))) v4i* as1_v4i_ptr;
typedef __attribute__((address_space(3))) v4i* as3_v4i_ptr;
#else
#define USE_ASYNC_LDS 0
#endif

// ---------------------------------------------------------------------------
// Helpers
// ---------------------------------------------------------------------------
__device__ __forceinline__ unsigned short f2bf(float f) {
  unsigned int u = __float_as_uint(f);
  u += 0x7FFFu + ((u >> 16) & 1u);          // round-to-nearest-even
  return (unsigned short)(u >> 16);
}

// A fragment: 16x32 (MxK), lane holds row m = lane%16, half = lane/16.
// Element e: K = (e<8 ? h*8+e : 16+h*8+(e-8))  -> two contiguous 16B loads.
__device__ __forceinline__ v16bf frag_a(const unsigned short* __restrict__ row,
                                        int kc, int h) {
  union { v16bf v; v8bf q[2]; } u;
  u.q[0] = *(const v8bf*)(row + kc + h * 8);
  u.q[1] = *(const v8bf*)(row + kc + 16 + h * 8);
  return u.v;
}

// B fragment: 32x16 (KxN), lane holds col n = lane%16, half = lane/16.
// Element e: K = (e<8 ? h*8+e : 16+h*8+(e-8))  -> two contiguous 16B loads.
__device__ __forceinline__ v16bf frag_b(const unsigned short* __restrict__ nrow,
                                        int kc, int h) {
  union { v16bf v; v8bf q[2]; } u;
  u.q[0] = *(const v8bf*)(nrow + kc + h * 8);
  u.q[1] = *(const v8bf*)(nrow + kc + 16 + h * 8);
  return u.v;
}

__device__ __forceinline__ v8f wmma_bf16(v16bf a, v16bf b, v8f c) {
  return __builtin_amdgcn_wmma_f32_16x16x32_bf16(
      /*neg_a=*/false, a, /*neg_b=*/false, b,
      /*c_mod=*/(short)0, c, /*reuse_a=*/false, /*reuse_b=*/false);
}

// ---------------------------------------------------------------------------
// K1a: x fp32 [bn][c][hw] -> bf16 transposed xbT [bn][hw][c]
// ---------------------------------------------------------------------------
__global__ void cvt_x_kernel(const float* __restrict__ x,
                             unsigned short* __restrict__ xbT) {
  const size_t total = (size_t)BN * C * HW;
  for (size_t i = (size_t)blockIdx.x * blockDim.x + threadIdx.x; i < total;
       i += (size_t)gridDim.x * blockDim.x) {
    int hw = (int)(i % HW);
    int c  = (int)((i / HW) % C);
    int bn = (int)(i / ((size_t)HW * C));
    xbT[((size_t)bn * HW + hw) * C + c] = f2bf(x[i]);
  }
}

// K1b: five weight matrices fp32 -> bf16 (layouts preserved: [o][c] / [c][p])
__global__ void cvt_w_kernel(const float* __restrict__ s0, const float* __restrict__ s1,
                             const float* __restrict__ s2, const float* __restrict__ s3,
                             const float* __restrict__ s4,
                             unsigned short* __restrict__ d0, unsigned short* __restrict__ d1,
                             unsigned short* __restrict__ d2, unsigned short* __restrict__ d3,
                             unsigned short* __restrict__ d4) {
  const float* srcs[5]          = {s0, s1, s2, s3, s4};
  unsigned short* dsts[5]       = {d0, d1, d2, d3, d4};
  const size_t per = (size_t)Pd * C;
  const size_t total = per * 5;
  for (size_t i = (size_t)blockIdx.x * blockDim.x + threadIdx.x; i < total;
       i += (size_t)gridDim.x * blockDim.x) {
    int sel = (int)(i / per);
    size_t off = i % per;
    dsts[sel][off] = f2bf(srcs[sel][off]);
  }
}

// ---------------------------------------------------------------------------
// K2: t/p/g = W[512x1024] @ X[1024x196] per image, bias fused.
//   t -> tT [b][l][p], p -> pT [b][m][p], g -> gpm [b][p][m] AND gT [b][l][p]
// grid (12, 13, 64): x = (which*4 + mgroup), y = ntile, z = bn; 8 waves/block
// Last n-tile: out-of-range columns load from a clamped address; their WMMA
// output columns are simply never stored (no exec-mask churn in the loop).
// ---------------------------------------------------------------------------
__global__ void gemm_tpg_kernel(const unsigned short* __restrict__ xbT,
                                const unsigned short* __restrict__ wt,
                                const unsigned short* __restrict__ wp,
                                const unsigned short* __restrict__ wg,
                                const float* __restrict__ t_bias,
                                const float* __restrict__ p_bias,
                                const float* __restrict__ g_bias,
                                unsigned short* __restrict__ tT,
                                unsigned short* __restrict__ pT,
                                unsigned short* __restrict__ gpm,
                                unsigned short* __restrict__ gT) {
  const int lane  = threadIdx.x & 31;
  const int wv    = threadIdx.x >> 5;
  const int which = blockIdx.x >> 2;                  // 0=t 1=p 2=g
  const int mtile = ((blockIdx.x & 3) << 3) + wv;     // 0..31
  const int ntile = blockIdx.y;                       // 0..12
  const int bn    = blockIdx.z;                       // 0..63
  const int b  = bn >> 3, nn = bn & 7;
  const int n  = lane & 15, h = lane >> 4;
  const int hw = ntile * 16 + n;
  const bool colok = (hw < HW);
  const int hwc = colok ? hw : (HW - 1);              // clamped load address

  const unsigned short* wmat = (which == 0) ? wt : (which == 1) ? wp : wg;
  const float* bias          = (which == 0) ? t_bias : (which == 1) ? p_bias : g_bias;
  const unsigned short* arow = wmat + (size_t)(mtile * 16 + n) * C;
  const unsigned short* xrow = xbT + ((size_t)bn * HW + hwc) * C;

  v8f acc; for (int r = 0; r < 8; ++r) acc[r] = 0.f;
  for (int kc = 0; kc < C; kc += 32)
    acc = wmma_bf16(frag_a(arow, kc, h), frag_b(xrow, kc, h), acc);

  if (!colok) return;                                  // last n-tile: cols 196..207
  const int o0 = mtile * 16 + h * 8;                   // output rows (P dim)
  const int lg = nn * HW + hw;                         // global l in [0,1568)

  v8u16 pk;
  for (int r = 0; r < 8; ++r) pk[r] = f2bf(acc[r] + bias[o0 + r]);

  // transposed store: [b][l][p], 8 contiguous bf16 = one 16B store
  unsigned short* dstT = (which == 0) ? tT : (which == 1) ? pT : gT;
  *(v8u16*)(dstT + ((size_t)(b * L + lg)) * Pd + o0) = pk;

  if (which == 2) {                                    // g also kept as [b][p][m]
    for (int r = 0; r < 8; ++r)
      gpm[((size_t)(b * Pd + o0 + r)) * L + lg] = pk[r];
  }
}

// ---------------------------------------------------------------------------
// K3: att[b][l][m] = sum_p tT[b][l][p] * pT[b][m][p]   (fp32 out)
// grid (98, 13, 8): block owns one l-tile; A-tile (16x512 bf16, 16KB) staged
// into LDS once via GLOBAL_LOAD_ASYNC_TO_LDS_B128 (ASYNCcnt), then 8 waves
// compute 8 m-tiles reading A-fragments from LDS (row pad 8 -> banks 4 apart).
// ---------------------------------------------------------------------------
constexpr int PADROW = Pd + 8;                  // 520 ushorts per LDS row

__global__ void gemm_att_kernel(const unsigned short* __restrict__ tT,
                                const unsigned short* __restrict__ pT,
                                float* __restrict__ att) {
  __shared__ unsigned short smA[16 * PADROW];   // 16.6 KB
  const int tid  = threadIdx.x;
  const int lane = tid & 31, wv = tid >> 5;
  const int ltile = blockIdx.x;                 // 0..97
  const int b     = blockIdx.z;

  // ---- stage A tile: 16 rows x 64 16B-chunks ----
  const unsigned short* srcA = tT + ((size_t)(b * L + ltile * 16)) * Pd;
  for (int q = tid; q < 16 * 64; q += 256) {
    const int r = q >> 6, c16 = q & 63;
    const unsigned short* g = srcA + (size_t)r * Pd + c16 * 8;
    unsigned short* d = smA + r * PADROW + c16 * 8;
#if USE_ASYNC_LDS
    __builtin_amdgcn_global_load_async_to_lds_b128(
        (as1_v4i_ptr)g, (as3_v4i_ptr)d, 0, 0);
#else
    *(v8u16*)d = *(const v8u16*)g;
#endif
  }
#if USE_ASYNC_LDS
  __builtin_amdgcn_s_wait_asynccnt(0);          // drain this wave's async copies
#endif
  __syncthreads();                              // all waves' data visible in LDS

  const int mtile = blockIdx.y * 8 + wv;
  if (mtile >= 98) return;                      // wave-uniform exit
  const int n = lane & 15, h = lane >> 4;

  const unsigned short* arow = smA + (size_t)n * PADROW;                  // LDS
  const unsigned short* brow = pT + ((size_t)(b * L + mtile * 16 + n)) * Pd;

  v8f acc; for (int r = 0; r < 8; ++r) acc[r] = 0.f;
  for (int kc = 0; kc < Pd; kc += 32) {
    __builtin_prefetch(brow + kc + 64, 0, 1);
    acc = wmma_bf16(frag_a(arow, kc, h), frag_b(brow, kc, h), acc);
  }
  const int l0 = ltile * 16 + h * 8;
  const int m  = mtile * 16 + n;
  float* dst = att + ((size_t)(b * L + l0)) * L + m;
  for (int r = 0; r < 8; ++r) dst[(size_t)r * L] = acc[r];
}

// ---------------------------------------------------------------------------
// K3s: row softmax over m; fp32 in, bf16 out. One block per (b,l) row.
// ---------------------------------------------------------------------------
__global__ void softmax_kernel(const float* __restrict__ att,
                               unsigned short* __restrict__ attb) {
  const int row = blockIdx.x;                      // 0 .. B*L-1
  const float* src = att + (size_t)row * L;
  unsigned short* dst = attb + (size_t)row * L;
  __shared__ float red[256];
  const int tid = threadIdx.x;

  float m = -3.4e38f;
  for (int i = tid; i < L; i += 256) m = fmaxf(m, src[i]);
  red[tid] = m; __syncthreads();
  for (int s = 128; s > 0; s >>= 1) {
    if (tid < s) red[tid] = fmaxf(red[tid], red[tid + s]);
    __syncthreads();
  }
  m = red[0]; __syncthreads();

  float sum = 0.f;
  for (int i = tid; i < L; i += 256) sum += __expf(src[i] - m);
  red[tid] = sum; __syncthreads();
  for (int s = 128; s > 0; s >>= 1) {
    if (tid < s) red[tid] += red[tid + s];
    __syncthreads();
  }
  const float inv = 1.f / red[0];
  for (int i = tid; i < L; i += 256) dst[i] = f2bf(__expf(src[i] - m) * inv);
}

// ---------------------------------------------------------------------------
// K4: x2[p][l] = sum_m g[p][m] * att[l][m]  -> store x2T [b][l][p] (bf16)
// grid (392, 8); 32x98 tiles, K = 1568 (49 chunks)
// ---------------------------------------------------------------------------
__global__ void gemm_x2_kernel(const unsigned short* __restrict__ gpm,
                               const unsigned short* __restrict__ attb,
                               unsigned short* __restrict__ x2T) {
  const int lane = threadIdx.x & 31, wv = threadIdx.x >> 5;
  const int tile = blockIdx.x * 8 + wv;            // 0..3135, exact
  const int b    = blockIdx.y;
  const int mtile = tile / 98, ntile = tile % 98;
  const int n = lane & 15, h = lane >> 4;

  const unsigned short* arow = gpm  + ((size_t)(b * Pd + mtile * 16 + n)) * L;
  const unsigned short* brow = attb + ((size_t)(b * L  + ntile * 16 + n)) * L;

  v8f acc; for (int r = 0; r < 8; ++r) acc[r] = 0.f;
  for (int kc = 0; kc < L; kc += 32) {
    __builtin_prefetch(arow + kc + 64, 0, 1);
    __builtin_prefetch(brow + kc + 64, 0, 1);
    acc = wmma_bf16(frag_a(arow, kc, h), frag_b(brow, kc, h), acc);
  }
  const int p0 = mtile * 16 + h * 8;
  const int lg = ntile * 16 + n;
  v8u16 pk;
  for (int r = 0; r < 8; ++r) pk[r] = f2bf(acc[r]);
  *(v8u16*)(x2T + ((size_t)(b * L + lg)) * Pd + p0) = pk;
}

// ---------------------------------------------------------------------------
// K5: y[c][l] = W1[c][:]·gT[l][:] + W2[c][:]·x2T[l][:] + b1[c] + b2[c]  (fp32)
// grid (784, 8); 64x98 tiles, K = 512, two WMMA per K-chunk into one acc
// ---------------------------------------------------------------------------
__global__ void gemm_y_kernel(const unsigned short* __restrict__ w1b,
                              const unsigned short* __restrict__ w2b,
                              const unsigned short* __restrict__ gT,
                              const unsigned short* __restrict__ x2T,
                              const float* __restrict__ bias1,
                              const float* __restrict__ bias2,
                              float* __restrict__ y) {
  const int lane = threadIdx.x & 31, wv = threadIdx.x >> 5;
  const int tile = blockIdx.x * 8 + wv;            // 0..6271, exact
  const int b    = blockIdx.y;
  const int mtile = tile / 98, ntile = tile % 98;
  const int n = lane & 15, h = lane >> 4;

  const unsigned short* a1 = w1b + (size_t)(mtile * 16 + n) * Pd;
  const unsigned short* a2 = w2b + (size_t)(mtile * 16 + n) * Pd;
  const unsigned short* b1 = gT  + ((size_t)(b * L + ntile * 16 + n)) * Pd;
  const unsigned short* b2 = x2T + ((size_t)(b * L + ntile * 16 + n)) * Pd;

  v8f acc; for (int r = 0; r < 8; ++r) acc[r] = 0.f;
  for (int kc = 0; kc < Pd; kc += 32) {
    acc = wmma_bf16(frag_a(a1, kc, h), frag_b(b1, kc, h), acc);
    acc = wmma_bf16(frag_a(a2, kc, h), frag_b(b2, kc, h), acc);
  }
  const int c0 = mtile * 16 + h * 8;
  const int lg = ntile * 16 + n;
  for (int r = 0; r < 8; ++r)
    y[((size_t)(b * C + c0 + r)) * L + lg] = acc[r] + bias1[c0 + r] + bias2[c0 + r];
}

// ---------------------------------------------------------------------------
// K6: BatchNorm batch stats per channel (mean, var over b,n,h,w)
// ---------------------------------------------------------------------------
__global__ void bn_stats_kernel(const float* __restrict__ y, float* __restrict__ stats) {
  const int c = blockIdx.x;
  __shared__ float rs[256], rs2[256];
  float s = 0.f, s2 = 0.f;
  for (int i = threadIdx.x; i < Bc * L; i += 256) {
    int b = i / L, l = i % L;
    float v = y[((size_t)(b * C + c)) * L + l];
    s += v; s2 += v * v;
  }
  rs[threadIdx.x] = s; rs2[threadIdx.x] = s2; __syncthreads();
  for (int t = 128; t > 0; t >>= 1) {
    if (threadIdx.x < t) { rs[threadIdx.x] += rs[threadIdx.x + t]; rs2[threadIdx.x] += rs2[threadIdx.x + t]; }
    __syncthreads();
  }
  if (threadIdx.x == 0) {
    const float inv_n = 1.f / (float)(Bc * L);
    float mean = rs[0] * inv_n;
    float var  = rs2[0] * inv_n - mean * mean;
    stats[2 * c]     = mean;
    stats[2 * c + 1] = var;
  }
}

// ---------------------------------------------------------------------------
// K7: out[bn][c][hw] = BN(y) + residual(x), back to [B*N, C, H, W] order
// ---------------------------------------------------------------------------
__global__ void finalize_kernel(const float* __restrict__ y, const float* __restrict__ x,
                                const float* __restrict__ stats,
                                const float* __restrict__ gamma,
                                const float* __restrict__ beta,
                                float* __restrict__ out) {
  const size_t total = (size_t)BN * C * HW;
  for (size_t i = (size_t)blockIdx.x * blockDim.x + threadIdx.x; i < total;
       i += (size_t)gridDim.x * blockDim.x) {
    int hw = (int)(i % HW);
    int c  = (int)((i / HW) % C);
    int bn = (int)(i / ((size_t)HW * C));
    int b = bn >> 3, nn = bn & 7;
    int l = nn * HW + hw;
    float v    = y[((size_t)(b * C + c)) * L + l];
    float mean = stats[2 * c], var = stats[2 * c + 1];
    out[i] = (v - mean) * rsqrtf(var + BN_EPS) * gamma[c] + beta[c] + x[i];
  }
}

// ---------------------------------------------------------------------------
// Launch
// ---------------------------------------------------------------------------
extern "C" void kernel_launch(void* const* d_in, const int* in_sizes, int n_in,
                              void* d_out, int out_size, void* d_ws, size_t ws_size,
                              hipStream_t stream) {
  const float* x     = (const float*)d_in[0];
  const float* t_w   = (const float*)d_in[1];
  const float* t_b   = (const float*)d_in[2];
  const float* p_w   = (const float*)d_in[3];
  const float* p_b   = (const float*)d_in[4];
  const float* g_w   = (const float*)d_in[5];
  const float* g_b   = (const float*)d_in[6];
  const float* w1_w  = (const float*)d_in[7];
  const float* w1_b  = (const float*)d_in[8];
  const float* w2_w  = (const float*)d_in[9];
  const float* w2_b  = (const float*)d_in[10];
  const float* gamma = (const float*)d_in[11];
  const float* beta  = (const float*)d_in[12];
  float* out = (float*)d_out;

  // workspace carve-out (all offsets 256B-aligned)
  char* ws = (char*)d_ws;
  size_t off = 0;
  auto take = [&](size_t bytes) { void* p = ws + off; off += (bytes + 255) & ~size_t(255); return p; };

  unsigned short* xbT  = (unsigned short*)take((size_t)BN * HW * C * 2);     // 25.7 MB
  unsigned short* wtb  = (unsigned short*)take((size_t)Pd * C * 2);
  unsigned short* wpb  = (unsigned short*)take((size_t)Pd * C * 2);
  unsigned short* wgb  = (unsigned short*)take((size_t)Pd * C * 2);
  unsigned short* w1bb = (unsigned short*)take((size_t)C * Pd * 2);
  unsigned short* w2bb = (unsigned short*)take((size_t)C * Pd * 2);
  unsigned short* tT   = (unsigned short*)take((size_t)Bc * L * Pd * 2);     // 12.8 MB
  unsigned short* pT   = (unsigned short*)take((size_t)Bc * L * Pd * 2);
  unsigned short* gpm  = (unsigned short*)take((size_t)Bc * Pd * L * 2);
  unsigned short* gT   = (unsigned short*)take((size_t)Bc * L * Pd * 2);
  unsigned short* x2T  = (unsigned short*)take((size_t)Bc * L * Pd * 2);
  unsigned short* attb = (unsigned short*)take((size_t)Bc * L * L * 2);      // 39.3 MB
  float*          stats = (float*)take((size_t)C * 2 * sizeof(float));
  // att_f (fp32, 78.7MB) overlaps y (fp32, 51.4MB): att_f dead after softmax
  float* att_f = (float*)take((size_t)Bc * L * L * sizeof(float));
  float* y     = att_f;

  // K1: conversions
  cvt_x_kernel<<<4096, 256, 0, stream>>>(x, xbT);
  cvt_w_kernel<<<1024, 256, 0, stream>>>(t_w, p_w, g_w, w1_w, w2_w,
                                         wtb, wpb, wgb, w1bb, w2bb);

  // K2: t/p/g projections (bias fused), dual-layout g
  gemm_tpg_kernel<<<dim3(12, 13, BN), 256, 0, stream>>>(
      xbT, wtb, wpb, wgb, t_b, p_b, g_b, tT, pT, gpm, gT);

  // K3: affinity  att = t^T p  (fp32), LDS-staged A tile (async copies)
  gemm_att_kernel<<<dim3(98, 13, Bc), 256, 0, stream>>>(tT, pT, att_f);

  // K3s: softmax over m, emit bf16
  softmax_kernel<<<Bc * L, 256, 0, stream>>>(att_f, attb);

  // K4: x2 = g @ att^T  -> x2T [b][l][p]
  gemm_x2_kernel<<<dim3(32 * 98 / 8, Bc), 256, 0, stream>>>(gpm, attb, x2T);

  // K5: y = W1*g + W2*x2 + biases (fp32, overlays att_f)
  gemm_y_kernel<<<dim3(64 * 98 / 8, Bc), 256, 0, stream>>>(
      w1bb, w2bb, gT, x2T, w1_b, w2_b, y);

  // K6/K7: BatchNorm + residual + layout restore
  bn_stats_kernel<<<C, 256, 0, stream>>>(y, stats);
  finalize_kernel<<<4096, 256, 0, stream>>>(y, x, stats, gamma, beta, out);
}
